// MultimodalBaseline_4810363372966
// MI455X (gfx1250) — matrix operand order repaired and checked by reading, not compiled, not run on hardware
//
#include <hip/hip_runtime.h>
#include <hip/hip_bf16.h>
#include <math.h>

typedef _Float16 half_t;
typedef __attribute__((ext_vector_type(8)))  _Float16 v8h;
typedef __attribute__((ext_vector_type(16))) _Float16 v16h;
typedef __attribute__((ext_vector_type(8)))  float    v8f;

#define D_MODEL 512
#define AS 520          // LDS row stride (halves) for 512-wide f16 buffers
#define HS 1040         // LDS row stride (halves) for 1024-wide FFN hidden
#define XS 516          // LDS row stride (floats) for f32 residual buffer

// Contiguous swizzled-weight block offsets (in halves). Each segment size is a
// multiple of 128 halves so the host-side allocator keeps them contiguous.
#define OFF_WQ 0
#define OFF_WK 262144
#define OFF_WV 524288
#define OFF_WO 786432
#define OFF_F1 1048576
#define OFF_F2 1572864
#define OFF_CG 2097152
// OFF_CB = OFF_CG + 262144; single-formula addressing handles both.

// ---------------- WMMA helpers (contiguous k order, k = 16*(lane/16) + elem) ----------------

__device__ __forceinline__ v8f wmma_f16(v16h a, v16h b, v8f c) {
  return __builtin_amdgcn_wmma_f32_16x16x32_f16(false, a, false, b, (short)0, c,
                                                false, false);
}

// A (16xK slice, row-major in LDS, stride `stride` halves):
// lane L<16: row=L, elems 0..15 -> k0..k0+15
// lane L>=16: row=L-16, elems 0..15 -> k0+16..k0+31
__device__ __forceinline__ v16h frag_a(const half_t* buf, int stride, int k0, int lane) {
  int row = lane & 15;
  int kb  = (lane & 16) ? 16 : 0;
  const half_t* p = buf + row * stride + k0 + kb;
  v8h lo = *(const v8h*)(p);
  v8h hi = *(const v8h*)(p + 8);
  return __builtin_shufflevector(lo, hi, 0,1,2,3,4,5,6,7,8,9,10,11,12,13,14,15);
}

// B from pre-swizzled global tile (1024B per tile, 32B per lane, coalesced)
__device__ __forceinline__ v16h frag_b_swz(const half_t* tile, int lane) {
  const half_t* p = tile + lane * 16;
  v8h lo = *(const v8h*)(p);
  v8h hi = *(const v8h*)(p + 8);
  return __builtin_shufflevector(lo, hi, 0,1,2,3,4,5,6,7,8,9,10,11,12,13,14,15);
}

// B from LDS row-major [n][k]: lane L<16 -> K 0..15, lane L>=16 -> K 16..31.
__device__ __forceinline__ v16h frag_b_lds(const half_t* buf, int stride, int k0, int lane) {
  int n  = lane & 15;
  int kk = (lane & 16);
  const half_t* p = buf + n * stride + k0 + kk;
  v8h lo = *(const v8h*)(p);
  v8h hi = *(const v8h*)(p + 8);
  return __builtin_shufflevector(lo, hi, 0,1,2,3,4,5,6,7,8,9,10,11,12,13,14,15);
}

// ---------------- kernel 1: weight swizzle (f32 [K][N] -> f16 B-tiles) -------

__global__ void swz_kernel(const float* __restrict__ src, half_t* __restrict__ dst,
                           int K, int N) {
  int t    = blockIdx.x * blockDim.x + threadIdx.x;
  int lane = t & 31;
  int tile = t >> 5;
  int ktiles = K >> 5;
  int ntiles = N >> 4;
  if (tile >= ktiles * ntiles) return;
  int nt = tile / ktiles;
  int kt = tile - nt * ktiles;
  int n     = nt * 16 + (lane & 15);
  int kbase = kt * 32 + ((lane & 16) ? 16 : 0);
  half_t* out = dst + (size_t)tile * 512 + lane * 16;
  #pragma unroll
  for (int i = 0; i < 16; ++i)
    out[i] = (half_t)src[(size_t)(kbase + i) * N + n];
}

// ---------------- kernel 2: FiLM gamma/beta = cond @ {cg_W, cb_W} ------------

__global__ __launch_bounds__(256)
void film_kernel(const float* __restrict__ cond,
                 const half_t* __restrict__ sCgCb,     // combined cg|cb tiles
                 const float* __restrict__ cg_b, const float* __restrict__ cb_b,
                 float* __restrict__ gbW, int Btot) {  // gamma | beta contiguous
  __shared__ __align__(16) half_t aF[16 * AS];
  int tid = threadIdx.x, row = tid >> 4, sub = tid & 15;
  int lane = tid & 31, wave = tid >> 5;
  int blk = blockIdx.x;

  for (int qq = 0; qq < 32; ++qq) {
    int c = sub + (qq << 4);
    aF[row * AS + c] = (half_t)cond[(size_t)(blk * 16 + row) * D_MODEL + c];
  }
  __syncthreads();

  v8f z = {0.f,0.f,0.f,0.f,0.f,0.f,0.f,0.f};
  v8f acc[8];
  #pragma unroll
  for (int m = 0; m < 8; ++m) acc[m] = z;

  for (int ks = 0; ks < 16; ++ks) {
    v16h a = frag_a(aF, AS, ks * 32, lane);
    // branch-free: tiles for nt in [0,64) are contiguous in sCgCb
    #pragma unroll
    for (int m = 0; m < 8; ++m) {
      int nt = wave + 8 * m;
      acc[m] = wmma_f16(a, frag_b_swz(sCgCb + (size_t)(nt * 16 + ks) * 512, lane),
                        acc[m]);
    }
  }

  int radd = (lane & 16) ? 8 : 0, nlo = lane & 15;
  #pragma unroll
  for (int m = 0; m < 8; ++m) {
    int nt = wave + 8 * m;
    int ncol = (nt & 31) * 16 + nlo;
    size_t base = (size_t)(nt >> 5) * (size_t)Btot * D_MODEL;   // 0: gamma, 1: beta
    float bia = (nt < 32) ? cg_b[ncol] : cb_b[ncol];
    #pragma unroll
    for (int jj = 0; jj < 8; ++jj) {
      int grow = blk * 16 + jj + radd;
      gbW[base + (size_t)grow * D_MODEL + ncol] = acc[m][jj] + bia;
    }
  }
}

// ---------------- kernel 3: fused attention + FFN, 1 WG per batch ------------

__device__ __forceinline__ void row_stats(const float* xbuf, float* red,
                                          float* rowstat, int row, int sub, int tid) {
  float s = 0.f, s2 = 0.f;
  #pragma unroll 8
  for (int qq = 0; qq < 32; ++qq) {
    int c = sub + (qq << 4);
    float v = xbuf[row * XS + c];
    s += v; s2 += v * v;
  }
  red[tid] = s; red[256 + tid] = s2;
  __syncthreads();
  if (sub == 0) {
    float a = 0.f, b2 = 0.f;
    #pragma unroll
    for (int ii = 0; ii < 16; ++ii) { a += red[row*16+ii]; b2 += red[256+row*16+ii]; }
    float mu  = a * (1.0f / 512.0f);
    float var = b2 * (1.0f / 512.0f) - mu * mu;
    rowstat[row]      = mu;
    rowstat[16 + row] = rsqrtf(var + 1e-5f);
  }
  __syncthreads();
}

__global__ __launch_bounds__(256)
void fused_kernel(const float* __restrict__ nf, const float* __restrict__ centers,
                  const float* __restrict__ pmaskg, const int* __restrict__ spkl,
                  const float* __restrict__ gamma, const float* __restrict__ beta,
                  const half_t* __restrict__ sW,          // unified weight base
                  const float* __restrict__ bq, const float* __restrict__ bk,
                  const float* __restrict__ bv, const float* __restrict__ bo,
                  const float* __restrict__ rW1, const float* __restrict__ rb1,
                  const float* __restrict__ rW2, const float* __restrict__ rb2,
                  const float* __restrict__ n1g, const float* __restrict__ n1b,
                  const float* __restrict__ cng, const float* __restrict__ cnb,
                  const float* __restrict__ f1b, const float* __restrict__ f2b,
                  const float* __restrict__ n2g, const float* __restrict__ n2b,
                  const float* __restrict__ spkb, float* __restrict__ out) {
  __shared__ __align__(16) float  xres[16 * XS];        // f32 residual (padded)
  __shared__ __align__(16) half_t aF[16 * AS];          // WMMA A staging
  __shared__ __align__(16) half_t qk[2 * 16 * AS];      // q | k, reused as FFN hidden
  __shared__ __align__(16) half_t vbuf[16 * AS];        // v
  __shared__ float red[2048];
  __shared__ float wmat[256];
  __shared__ float rbias[256];
  __shared__ float cc[32];
  __shared__ float pmask[16];
  __shared__ float rowstat[32];
  __shared__ float sbv[3];

  const int b = blockIdx.x;
  const int tid = threadIdx.x, row = tid >> 4, sub = tid & 15;
  const int lane = tid & 31, wave = tid >> 5;
  const int radd = (lane & 16) ? 8 : 0, nlo = lane & 15;
  const v8f vz = {0.f,0.f,0.f,0.f,0.f,0.f,0.f,0.f};

  // ---- stage A: load, LayerNorm(cn), FiLM ----
  if (tid < 16) {
    cc[tid]      = centers[(size_t)(b * 16 + tid) * 2 + 0];
    cc[16 + tid] = centers[(size_t)(b * 16 + tid) * 2 + 1];
    pmask[tid]   = pmaskg[(size_t)b * 16 + tid];
  }
  if (tid < 3) sbv[tid] = spkb[tid];
  const float* nfb = nf + (size_t)b * 16 * D_MODEL;
  for (int qq = 0; qq < 32; ++qq) {
    int c = sub + (qq << 4);
    xres[row * XS + c] = nfb[row * D_MODEL + c];
  }
  __syncthreads();
  row_stats(xres, red, rowstat, row, sub, tid);
  {
    const float* gR = gamma + (size_t)b * D_MODEL;
    const float* bR = beta  + (size_t)b * D_MODEL;
    float mu = rowstat[row], rs = rowstat[16 + row];
    for (int qq = 0; qq < 32; ++qq) {
      int c = sub + (qq << 4);
      float v  = xres[row * XS + c];
      float xl = (v - mu) * rs * cng[c] + cnb[c];
      float xf = xl * (1.0f + gR[c]) + bR[c];
      xres[row * XS + c] = xf;
      aF[row * AS + c] = (half_t)xf;
    }
  }
  __syncthreads();

  // ---- stage B: rel-position MLP bias (VALU, float4-vectorized) ----
  {
    float dx = cc[row] - cc[sub], dy = cc[16 + row] - cc[16 + sub];
    const float4* w1a = (const float4*)rW1;              // rows 0 / 1 of rel_W1
    const float4* w1b = (const float4*)(rW1 + D_MODEL);
    const float4* b1v = (const float4*)rb1;
    const float4* w2v = (const float4*)rW2;
    float acc = 0.f;
    for (int d4 = 0; d4 < D_MODEL / 4; ++d4) {
      float4 ax = w1a[d4], ay = w1b[d4], ab = b1v[d4], aw = w2v[d4];
      float h0 = fmaxf(fmaf(dx, ax.x, fmaf(dy, ay.x, ab.x)), 0.f);
      float h1 = fmaxf(fmaf(dx, ax.y, fmaf(dy, ay.y, ab.y)), 0.f);
      float h2 = fmaxf(fmaf(dx, ax.z, fmaf(dy, ay.z, ab.z)), 0.f);
      float h3 = fmaxf(fmaf(dx, ax.w, fmaf(dy, ay.w, ab.w)), 0.f);
      acc = fmaf(h0, aw.x, fmaf(h1, aw.y, fmaf(h2, aw.z, fmaf(h3, aw.w, acc))));
    }
    rbias[tid] = tanhf(acc + rb2[0]) * 2.0f;
  }

  // ---- stage C: Q,K,V GEMMs (12 tiles/wave x 16 k-steps, WMMA f16) ----
  {
    v8f acc[12];
    #pragma unroll
    for (int u = 0; u < 12; ++u) acc[u] = vz;
    for (int ks = 0; ks < 16; ++ks) {
      v16h a = frag_a(aF, AS, ks * 32, lane);
      #pragma unroll
      for (int m = 0; m < 4; ++m) {
        int nt = wave + 8 * m;
        size_t tb = (size_t)(nt * 16 + ks) * 512;
        // batch the three B-fragment loads so waits can be partial
        v16h bqf = frag_b_swz(sW + OFF_WQ + tb, lane);
        v16h bkf = frag_b_swz(sW + OFF_WK + tb, lane);
        v16h bvf = frag_b_swz(sW + OFF_WV + tb, lane);
        acc[m]     = wmma_f16(a, bqf, acc[m]);
        acc[4 + m] = wmma_f16(a, bkf, acc[4 + m]);
        acc[8 + m] = wmma_f16(a, bvf, acc[8 + m]);
      }
    }
    #pragma unroll
    for (int m = 0; m < 4; ++m) {
      int col = (wave + 8 * m) * 16 + nlo;
      float cq = bq[col], ck = bk[col], cv = bv[col];
      #pragma unroll
      for (int jj = 0; jj < 8; ++jj) {
        int r = jj + radd;
        qk[r * AS + col]            = (half_t)(acc[m][jj] + cq);
        qk[16 * AS + r * AS + col]  = (half_t)(acc[4 + m][jj] + ck);
        vbuf[r * AS + col]          = (half_t)(acc[8 + m][jj] + cv);
      }
    }
  }
  __syncthreads();

  // ---- stage D: logits = q @ k^T, K split across 8 waves ----
  {
    v8f lc = vz;
    int k0 = wave * 64;
    lc = wmma_f16(frag_a(qk, AS, k0, lane),      frag_b_lds(qk + 16 * AS, AS, k0, lane),      lc);
    lc = wmma_f16(frag_a(qk, AS, k0 + 32, lane), frag_b_lds(qk + 16 * AS, AS, k0 + 32, lane), lc);
    #pragma unroll
    for (int jj = 0; jj < 8; ++jj)
      red[wave * 256 + (jj + radd) * 16 + nlo] = lc[jj];
  }
  __syncthreads();

  // ---- stage E: bias, mask, softmax over 16 cols ----
  {
    int spk = spkl[b];
    float lg = 0.f;
    #pragma unroll
    for (int w = 0; w < 8; ++w) lg += red[w * 256 + tid];
    lg *= 0.04419417382415922f;                      // 1/sqrt(512)
    int et = (sub == spk) ? 2 : ((row == spk) ? 1 : 0);
    lg += rbias[tid] + sbv[et];
    bool validj = pmask[sub] > 0.5f;
    if (row == sub || !validj) lg = -10000.0f;
    wmat[tid] = lg;
    __syncthreads();
    if (sub == 0) {
      float m = -1e30f;
      #pragma unroll
      for (int jj = 0; jj < 16; ++jj) m = fmaxf(m, wmat[row * 16 + jj]);
      rowstat[row] = m;
    }
    __syncthreads();
    float e = __expf(wmat[tid] - rowstat[row]);
    wmat[tid] = e;
    __syncthreads();
    if (sub == 0) {
      float s = 0.f;
      #pragma unroll
      for (int jj = 0; jj < 16; ++jj) s += wmat[row * 16 + jj];
      rowstat[16 + row] = s;
    }
    __syncthreads();
    float wv = (wmat[tid] / rowstat[16 + row]) * (validj ? 1.f : 0.f);
    wmat[tid] = wv;
  }
  __syncthreads();

  // ---- stage F: agg = w @ v (tiny, VALU with 16B LDS loads), row-mask ----
  {
    float wr[16];
    #pragma unroll
    for (int jj = 0; jj < 16; ++jj) wr[jj] = wmat[row * 16 + jj];
    float pm = pmask[row];
    #pragma unroll
    for (int qq = 0; qq < 4; ++qq) {
      int c0 = (sub << 5) + (qq << 3);               // 8-col chunk
      float va[8];
      #pragma unroll
      for (int e = 0; e < 8; ++e) va[e] = 0.f;
      #pragma unroll
      for (int jj = 0; jj < 16; ++jj) {
        v8h vv = *(const v8h*)&vbuf[jj * AS + c0];
        #pragma unroll
        for (int e = 0; e < 8; ++e)
          va[e] = fmaf(wr[jj], (float)vv[e], va[e]);
      }
      v8h hv;
      #pragma unroll
      for (int e = 0; e < 8; ++e) hv[e] = (half_t)(va[e] * pm);
      *(v8h*)&aF[row * AS + c0] = hv;
    }
  }
  __syncthreads();

  // ---- stage G: out-proj Wo + residual ----
  {
    v8f acc[4];
    #pragma unroll
    for (int u = 0; u < 4; ++u) acc[u] = vz;
    for (int ks = 0; ks < 16; ++ks) {
      v16h a = frag_a(aF, AS, ks * 32, lane);
      #pragma unroll
      for (int m = 0; m < 4; ++m) {
        int nt = wave + 8 * m;
        acc[m] = wmma_f16(a, frag_b_swz(sW + OFF_WO + (size_t)(nt * 16 + ks) * 512, lane),
                          acc[m]);
      }
    }
    #pragma unroll
    for (int m = 0; m < 4; ++m) {
      int col = (wave + 8 * m) * 16 + nlo;
      float bia = bo[col];
      #pragma unroll
      for (int jj = 0; jj < 8; ++jj) {
        int idx = (jj + radd) * XS + col;
        xres[idx] = xres[idx] + acc[m][jj] + bia;
      }
    }
  }
  __syncthreads();

  // ---- stage H: LayerNorm n1 ----
  row_stats(xres, red, rowstat, row, sub, tid);
  {
    float mu = rowstat[row], rs = rowstat[16 + row];
    for (int qq = 0; qq < 32; ++qq) {
      int c = sub + (qq << 4);
      float xl = (xres[row * XS + c] - mu) * rs * n1g[c] + n1b[c];
      xres[row * XS + c] = xl;
      aF[row * AS + c] = (half_t)xl;
    }
  }
  __syncthreads();

  // ---- stage I: FFN up 512->1024 + ReLU ----
  {
    v8f acc[8];
    #pragma unroll
    for (int u = 0; u < 8; ++u) acc[u] = vz;
    for (int ks = 0; ks < 16; ++ks) {
      v16h a = frag_a(aF, AS, ks * 32, lane);
      #pragma unroll
      for (int m = 0; m < 8; m += 2) {
        int nt0 = wave + 8 * m, nt1 = wave + 8 * (m + 1);
        v16h b0 = frag_b_swz(sW + OFF_F1 + (size_t)(nt0 * 16 + ks) * 512, lane);
        v16h b1 = frag_b_swz(sW + OFF_F1 + (size_t)(nt1 * 16 + ks) * 512, lane);
        acc[m]     = wmma_f16(a, b0, acc[m]);
        acc[m + 1] = wmma_f16(a, b1, acc[m + 1]);
      }
    }
    #pragma unroll
    for (int m = 0; m < 8; ++m) {
      int col = (wave + 8 * m) * 16 + nlo;     // 0..1023
      float bia = f1b[col];
      #pragma unroll
      for (int jj = 0; jj < 8; ++jj)
        qk[(jj + radd) * HS + col] = (half_t)fmaxf(acc[m][jj] + bia, 0.f);
    }
  }
  __syncthreads();

  // ---- stage J: FFN down 1024->512 + residual ----
  {
    v8f acc[4];
    #pragma unroll
    for (int u = 0; u < 4; ++u) acc[u] = vz;
    for (int ks = 0; ks < 32; ++ks) {
      v16h a = frag_a(qk, HS, ks * 32, lane);
      #pragma unroll
      for (int m = 0; m < 4; m += 2) {
        int nt0 = wave + 8 * m, nt1 = wave + 8 * (m + 1);
        v16h b0 = frag_b_swz(sW + OFF_F2 + (size_t)(nt0 * 32 + ks) * 512, lane);
        v16h b1 = frag_b_swz(sW + OFF_F2 + (size_t)(nt1 * 32 + ks) * 512, lane);
        acc[m]     = wmma_f16(a, b0, acc[m]);
        acc[m + 1] = wmma_f16(a, b1, acc[m + 1]);
      }
    }
    #pragma unroll
    for (int m = 0; m < 4; ++m) {
      int col = (wave + 8 * m) * 16 + nlo;
      float bia = f2b[col];
      #pragma unroll
      for (int jj = 0; jj < 8; ++jj) {
        int idx = (jj + radd) * XS + col;
        xres[idx] = xres[idx] + acc[m][jj] + bia;
      }
    }
  }
  __syncthreads();

  // ---- stage K: LayerNorm n2, output mask, store ----
  row_stats(xres, red, rowstat, row, sub, tid);
  {
    float mu = rowstat[row], rs = rowstat[16 + row];
    float pm = pmask[row];
    float* ob = out + (size_t)b * 16 * D_MODEL;
    for (int qq = 0; qq < 32; ++qq) {
      int c = sub + (qq << 4);
      float xl = (xres[row * XS + c] - mu) * rs * n2g[c] + n2b[c];
      ob[row * D_MODEL + c] = xl * pm;
    }
  }
}

// ---------------- host launch -----------------------------------------------

extern "C" void kernel_launch(void* const* d_in, const int* in_sizes, int n_in,
                              void* d_out, int out_size, void* d_ws, size_t ws_size,
                              hipStream_t stream) {
  (void)n_in; (void)out_size; (void)ws_size;
  const float* nf      = (const float*)d_in[0];
  const float* centers = (const float*)d_in[1];
  const float* pmaskg  = (const float*)d_in[2];
  const int*   spkl    = (const int*)d_in[3];
  const float* cond    = (const float*)d_in[4];
  const float* Wq = (const float*)d_in[5];   const float* bq = (const float*)d_in[6];
  const float* Wk = (const float*)d_in[7];   const float* bk = (const float*)d_in[8];
  const float* Wv = (const float*)d_in[9];   const float* bv = (const float*)d_in[10];
  const float* rW1 = (const float*)d_in[11]; const float* rb1 = (const float*)d_in[12];
  const float* rW2 = (const float*)d_in[13]; const float* rb2 = (const float*)d_in[14];
  const float* Wo = (const float*)d_in[15];  const float* bo = (const float*)d_in[16];
  const float* n1g = (const float*)d_in[17]; const float* n1b = (const float*)d_in[18];
  const float* cng = (const float*)d_in[19]; const float* cnb = (const float*)d_in[20];
  const float* cgW = (const float*)d_in[21]; const float* cgb = (const float*)d_in[22];
  const float* cbW = (const float*)d_in[23]; const float* cbb = (const float*)d_in[24];
  const float* f1W = (const float*)d_in[25]; const float* f1b = (const float*)d_in[26];
  const float* f2W = (const float*)d_in[27]; const float* f2b = (const float*)d_in[28];
  const float* n2g = (const float*)d_in[29]; const float* n2b = (const float*)d_in[30];
  const float* spkb = (const float*)d_in[31];
  float* out = (float*)d_out;

  const int B = in_sizes[4] / D_MODEL;       // 2048

  char* ws = (char*)d_ws;
  // one contiguous block for all swizzled weights (offsets match OFF_* above)
  half_t* wall = (half_t*)ws;                               // 2,621,440 halves
  float*  gbW  = (float*)(ws + (size_t)2621440 * 2 + 256);  // gamma | beta
  gbW = (float*)(((uintptr_t)gbW + 255) & ~(uintptr_t)255);

  struct Job { const float* s; size_t off; int K; int N; };
  Job jobs[8] = {
    {Wq,  OFF_WQ, 512, 512}, {Wk, OFF_WK, 512, 512}, {Wv, OFF_WV, 512, 512},
    {Wo,  OFF_WO, 512, 512}, {f1W, OFF_F1, 512, 1024}, {f2W, OFF_F2, 1024, 512},
    {cgW, OFF_CG, 512, 512}, {cbW, OFF_CG + 262144, 512, 512},
  };
  for (int j = 0; j < 8; ++j) {
    int tiles = (jobs[j].K >> 5) * (jobs[j].N >> 4);
    int thr = tiles * 32;
    swz_kernel<<<(thr + 255) / 256, 256, 0, stream>>>(jobs[j].s, wall + jobs[j].off,
                                                      jobs[j].K, jobs[j].N);
  }

  film_kernel<<<B / 16, 256, 0, stream>>>(cond, wall + OFF_CG, cgb, cbb, gbW, B);

  const float* gammaW = gbW;
  const float* betaW  = gbW + (size_t)B * D_MODEL;
  fused_kernel<<<B, 256, 0, stream>>>(nf, centers, pmaskg, spkl, gammaW, betaW,
                                      wall,
                                      bq, bk, bv, bo, rW1, rb1, rW2, rb2,
                                      n1g, n1b, cng, cnb, f1b, f2b, n2g, n2b,
                                      spkb, out);
}
